// Attention_mydecoder_43035572306559
// MI455X (gfx1250) — matrix operand order repaired and checked by reading, not compiled, not run on hardware
//
#include <hip/hip_runtime.h>
#include <hip/hip_bf16.h>

// ---------------------------------------------------------------------------
// Cross-attention (B=8, N_TXT=512, N_IMG=4096, DIM=1024, H=16, DH=64)
// bf16 WMMA pipeline on gfx1250 (wave32):
//   - activations pre-converted to bf16 so every GEMM tile is a pure
//     GLOBAL_LOAD_ASYNC_TO_LDS_B128 copy (no VALU in the hot loop)
//   - 256x128 blocks, 64x64 wave tiles: 16 WMMA per 16 ds_load_b128
//   - double-buffered LDS, one barrier per 64-deep K tile
//   - V output tile re-transposed through LDS, stored with
//     GLOBAL_STORE_ASYNC_FROM_LDS_B128 (256B contiguous per thread)
//   - flash attention with 128-key blocks (halved softmax/barrier overhead)
// ---------------------------------------------------------------------------

typedef __attribute__((ext_vector_type(16))) __bf16 v16bf;
typedef __attribute__((ext_vector_type(2)))  __bf16 v2bf;
typedef __attribute__((ext_vector_type(8)))  float  v8f;

#define LDT  80    // LDS row stride for 64-wide WMMA tiles (160B)
#define LDP  136   // LDS row stride for 128-wide P / V^T tiles (272B)
#define LDVS 264   // LDS row stride for 256-wide V transpose scratch (528B)

static __device__ __forceinline__ unsigned int bf16_1(float f) {
  unsigned u = __float_as_uint(f);
  unsigned r = 0x7FFFu + ((u >> 16) & 1u);           // round-to-nearest-even
  return (u + r) >> 16;
}
static __device__ __forceinline__ unsigned int packbf2(float a, float b) {
#if __has_builtin(__builtin_amdgcn_cvt_pk_bf16_f32)
  v2bf r = __builtin_amdgcn_cvt_pk_bf16_f32(a, b);
  return __builtin_bit_cast(unsigned, r);
#else
  return bf16_1(a) | (bf16_1(b) << 16);
#endif
}
static __device__ __forceinline__ unsigned short bf16s(float f) {
  return (unsigned short)packbf2(f, f);
}

// Async copies, tracked by ASYNCcnt. GVS mode: uniform SGPR64 base + 32-bit voffset.
static __device__ __forceinline__ void async_b128(const void* gbase, unsigned byte_off,
                                                  void* lds_dst) {
  asm volatile("global_load_async_to_lds_b128 %0, %1, %2"
               :: "v"((unsigned)(uintptr_t)lds_dst), "v"(byte_off), "s"(gbase)
               : "memory");
}
static __device__ __forceinline__ void async_store_b128(const void* gbase, unsigned byte_off,
                                                        const void* lds_src) {
  asm volatile("global_store_async_from_lds_b128 %0, %1, %2"
               :: "v"(byte_off), "v"((unsigned)(uintptr_t)lds_src), "s"(gbase)
               : "memory");
}
static __device__ __forceinline__ void wait_async0() {
  asm volatile("s_wait_asynccnt 0" ::: "memory");
}

union FragBF { uint4 q[2]; v16bf v; };

// A-frag 16x32 bf16: lane row = lane&15; kb = (lane>>4)*8; elems = K[kb..+7], K[kb+16..+23]
static __device__ __forceinline__ v16bf load_a_frag(const unsigned short* lds, int row0,
                                                    int kofs, int lane, int ld) {
  int row = row0 + (lane & 15);
  int kb  = kofs + ((lane >> 4) << 3);
  const unsigned short* p = lds + row * ld + kb;
  FragBF f;
  f.q[0] = *(const uint4*)(p);
  f.q[1] = *(const uint4*)(p + 16);
  return f.v;
}
// B-frag 32x16 bf16 from [n][k] LDS: lane col = lane&15; kb = (lane>>4)*8; elems = K[kb..+7], K[kb+16..+23]
static __device__ __forceinline__ v16bf load_b_frag(const unsigned short* lds, int col0,
                                                    int kofs, int lane, int ld) {
  int col = col0 + (lane & 15);
  int kb  = kofs + ((lane >> 4) << 3);
  const unsigned short* p = lds + col * ld + kb;
  FragBF f;
  f.q[0] = *(const uint4*)(p);
  f.q[1] = *(const uint4*)(p + 16);
  return f.v;
}

// ---------------------------------------------------------------------------
// 256x128 block GEMM, 256 threads = 8 waves (4M x 2N), wave tile 64x64.
// A (bf16 [M][K]) and B (bf16 [N][K]) tiles are pure async copies,
// double-buffered, one barrier per 64-deep K tile.
// "fin" runs after a barrier; may use the A-tile LDS (80KB) as scratch.
// ---------------------------------------------------------------------------
template <typename Fin>
static __device__ __forceinline__ void gemm256x128(const unsigned short* Abf,
                                                   const unsigned short* Bt,
                                                   int Kg, Fin fin) {
  __shared__ __align__(16) unsigned short As[2][256 * LDT];
  __shared__ __align__(16) unsigned short Bs[2][128 * LDT];
  const int tid  = threadIdx.x;
  const int lane = tid & 31, wid = tid >> 5;
  const int wm = (wid & 3) * 64, wn = (wid >> 2) * 64;
  const int m0 = blockIdx.y * 256, n0 = blockIdx.x * 128;

  v8f acc[4][4];
#pragma unroll
  for (int i = 0; i < 4; i++)
#pragma unroll
    for (int j = 0; j < 4; j++) acc[i][j] = {};

  // A fill: one 64-col row per thread (8 x b128). B fill: half row (4 x b128).
  const unsigned short* Abase = Abf + (size_t)m0 * Kg;
  const unsigned boffA = (unsigned)tid * Kg * 2;
  const int lrB = tid >> 1, lcB = (tid & 1) * 32;
  const unsigned short* Bbase = Bt + (size_t)n0 * Kg;
  const unsigned boffB = ((unsigned)lrB * Kg + lcB) * 2;

  // ---- prolog: tile 0 in flight ----
#pragma unroll
  for (int i = 0; i < 8; i++)
    async_b128(Abase, boffA + i * 16, &As[0][tid * LDT + i * 8]);
#pragma unroll
  for (int i = 0; i < 4; i++)
    async_b128(Bbase, boffB + i * 16, &Bs[0][lrB * LDT + lcB + i * 8]);

  int p = 0;
  for (int k0 = 0; k0 < Kg; k0 += 64, p ^= 1) {
    wait_async0();
    __syncthreads();
    if (k0 + 64 < Kg) {  // prefetch tile i+1 (overlaps the WMMAs below)
#pragma unroll
      for (int i = 0; i < 8; i++)
        async_b128(Abase, boffA + (unsigned)(k0 + 64) * 2 + i * 16,
                   &As[p ^ 1][tid * LDT + i * 8]);
#pragma unroll
      for (int i = 0; i < 4; i++)
        async_b128(Bbase, boffB + (unsigned)(k0 + 64) * 2 + i * 16,
                   &Bs[p ^ 1][lrB * LDT + lcB + i * 8]);
    }
#pragma unroll
    for (int ks = 0; ks < 2; ks++) {
      v16bf a[4], b[4];
#pragma unroll
      for (int mi = 0; mi < 4; mi++) a[mi] = load_a_frag(As[p], wm + mi * 16, ks * 32, lane, LDT);
#pragma unroll
      for (int ni = 0; ni < 4; ni++) b[ni] = load_b_frag(Bs[p], wn + ni * 16, ks * 32, lane, LDT);
#pragma unroll
      for (int ni = 0; ni < 4; ni++)
#pragma unroll
        for (int mi = 0; mi < 4; mi++)
          acc[mi][ni] = __builtin_amdgcn_wmma_f32_16x16x32_bf16(false, a[mi], false, b[ni],
                                                                (short)0, acc[mi][ni], false, false);
    }
  }
  __syncthreads();  // LDS tile buffers now reusable as scratch
  fin(acc, m0, n0, wm, wn, lane, tid, &As[0][0]);
}

// ---------------------------------------------------------------------------
// Weight transpose + fp32->bf16: src [K][N] -> dst [N][K]
// ---------------------------------------------------------------------------
__global__ void __launch_bounds__(256) transpose_cvt_kernel(const float* __restrict__ src,
                                                            unsigned short* __restrict__ dst,
                                                            int K, int N) {
  __shared__ float tile[32][33];
  const int x = threadIdx.x & 31;
  const int y = threadIdx.x >> 5;  // 0..7
  const int kb = blockIdx.y << 5, nb = blockIdx.x << 5;
#pragma unroll
  for (int j = 0; j < 4; j++)
    tile[y + 8 * j][x] = src[(size_t)(kb + y + 8 * j) * N + nb + x];
  __syncthreads();
#pragma unroll
  for (int j = 0; j < 4; j++)
    dst[(size_t)(nb + y + 8 * j) * K + kb + x] = bf16s(tile[x][y + 8 * j]);
}

// Elementwise fp32 -> bf16 (8 elems / thread, 16B stores)
__global__ void __launch_bounds__(256) cvt_bf16_kernel(const float* __restrict__ src,
                                                       unsigned short* __restrict__ dst) {
  const size_t i = ((size_t)blockIdx.x * 256 + threadIdx.x) * 8;
  float4 f0 = *(const float4*)(src + i);
  float4 f1 = *(const float4*)(src + i + 4);
  unsigned u[4] = { packbf2(f0.x, f0.y), packbf2(f0.z, f0.w),
                    packbf2(f1.x, f1.y), packbf2(f1.z, f1.w) };
  *(uint4*)(dst + i) = *(uint4*)u;
}

// ---------------------------------------------------------------------------
// GEMM kernels
// ---------------------------------------------------------------------------
__global__ void __launch_bounds__(256) gemm_q_kernel(const unsigned short* __restrict__ txt16,
                                                     const unsigned short* __restrict__ WqT,
                                                     unsigned short* __restrict__ q_ws) {
  gemm256x128(txt16, WqT, 1024,
    [=](v8f (&acc)[4][4], int m0, int n0, int wm, int wn, int lane, int tid,
        unsigned short* scratch) {
      (void)tid; (void)scratch;
      const int half = lane >> 4, nl = lane & 15;
#pragma unroll
      for (int mi = 0; mi < 4; mi++)
#pragma unroll
        for (int ni = 0; ni < 4; ni++)
#pragma unroll
          for (int r = 0; r < 8; r++) {
            int m = m0 + wm + mi * 16 + r + 8 * half;
            int n = n0 + wn + ni * 16 + nl;
            int b = m >> 9, i = m & 511, h = n >> 6, d = n & 63;
            q_ws[((((size_t)b * 16 + h) * 512 + i) << 6) + d] =
                bf16s(acc[mi][ni][r] * 0.125f);
          }
    });
}

__global__ void __launch_bounds__(256) gemm_kv_kernel(const unsigned short* __restrict__ img16,
                                                      const unsigned short* __restrict__ WkvT,
                                                      unsigned short* __restrict__ k_ws,
                                                      unsigned short* __restrict__ v_ws) {
  gemm256x128(img16, WkvT, 1024,
    [=](v8f (&acc)[4][4], int m0, int n0, int wm, int wn, int lane, int tid,
        unsigned short* scratch) {
      const int half = lane >> 4, nl = lane & 15;
      if (n0 < 1024) {
        // K half: [b,h,j,d] — d contiguous across lanes
#pragma unroll
        for (int mi = 0; mi < 4; mi++)
#pragma unroll
          for (int ni = 0; ni < 4; ni++)
#pragma unroll
            for (int r = 0; r < 8; r++) {
              int m = m0 + wm + mi * 16 + r + 8 * half;
              int n = n0 + wn + ni * 16 + nl;
              int b = m >> 12, j = m & 4095, h = n >> 6, d = n & 63;
              k_ws[((((size_t)b * 16 + h) * 4096 + j) << 6) + d] = bf16s(acc[mi][ni][r]);
            }
      } else {
        // V half: transpose 256x128 tile in LDS, async-store contiguous rows
#pragma unroll
        for (int mi = 0; mi < 4; mi++)
#pragma unroll
          for (int ni = 0; ni < 4; ni++)
#pragma unroll
            for (int r = 0; r < 8; r++) {
              int nloc = wn + ni * 16 + nl;
              int mloc = wm + mi * 16 + r + 8 * half;
              scratch[nloc * LDVS + mloc] = bf16s(acc[mi][ni][r]);
            }
        __syncthreads();
        const int nr = tid >> 1;            // 0..127: local n row
        const int mh = (tid & 1) * 128;     // half of the 256 m's
        const int n2 = (n0 - 1024) + nr;    // flat h*64+d == n2
        const int b  = m0 >> 12;
        const int j0 = (m0 & 4095) + mh;
        const unsigned short* src = scratch + nr * LDVS + mh;
        unsigned short* vb = v_ws + (size_t)b * 16 * 64 * 4096;   // uniform base
        const unsigned off = ((unsigned)n2 * 4096 + j0) * 2;
#pragma unroll
        for (int i = 0; i < 16; i++)
          async_store_b128(vb, off + i * 16, src + i * 8);
        // ASYNCcnt drained by S_ENDPGM's implicit wait-idle
      }
    });
}

__global__ void __launch_bounds__(256) gemm_out_kernel(const unsigned short* __restrict__ o_ws,
                                                       const unsigned short* __restrict__ WoutT,
                                                       const float* __restrict__ bout,
                                                       float* __restrict__ out) {
  gemm256x128(o_ws, WoutT, 1024,
    [=](v8f (&acc)[4][4], int m0, int n0, int wm, int wn, int lane, int tid,
        unsigned short* scratch) {
      (void)tid; (void)scratch;
      const int half = lane >> 4, nl = lane & 15;
#pragma unroll
      for (int mi = 0; mi < 4; mi++)
#pragma unroll
        for (int ni = 0; ni < 4; ni++)
#pragma unroll
          for (int r = 0; r < 8; r++) {
            int m = m0 + wm + mi * 16 + r + 8 * half;
            int n = n0 + wn + ni * 16 + nl;
            out[(size_t)m * 1024 + n] = acc[mi][ni][r] + bout[n];
          }
    });
}

// ---------------------------------------------------------------------------
// Flash attention: block = (b, h, 128-query chunk); 8 waves x 16 queries.
// 128-key blocks, double-buffered async K/V tiles, one main barrier per block.
// ---------------------------------------------------------------------------
__global__ void __launch_bounds__(256) attn_kernel(const unsigned short* __restrict__ q_ws,
                                                   const unsigned short* __restrict__ k_ws,
                                                   const unsigned short* __restrict__ v_ws,
                                                   const unsigned char* __restrict__ q_mask,
                                                   const unsigned char* __restrict__ kv_mask,
                                                   unsigned short* __restrict__ o_ws) {
  __shared__ __align__(16) unsigned short Ks[2][128 * LDT];   // [key][d]
  __shared__ __align__(16) unsigned short Vts[2][64 * LDP];   // [d][key]
  __shared__ __align__(16) unsigned short Ps[8 * 16 * LDP];   // per-wave P scratch

  const int tid = threadIdx.x, lane = tid & 31, wid = tid >> 5;
  const int half = lane >> 4, nl = lane & 15;
  const int bh = blockIdx.x >> 2;              // b*16+h
  const int qc = blockIdx.x & 3;
  const int b  = bh >> 4, h = bh & 15;
  const int q0 = qc * 128 + wid * 16;          // wave's query base (within 512)

  // Q fragments (A layout), prescaled by d^-0.5 in GEMM1
  v16bf qf[2];
  {
    const unsigned short* qp = q_ws + (((size_t)bh * 512 + q0 + nl) << 6);
    const int kb = half << 3;
#pragma unroll
    for (int c = 0; c < 2; c++) {
      FragBF f;
      f.q[0] = *(const uint4*)(qp + c * 32 + kb);
      f.q[1] = *(const uint4*)(qp + c * 32 + kb + 16);
      qf[c] = f.v;
    }
  }
  float qm[8];
#pragma unroll
  for (int r = 0; r < 8; r++)
    qm[r] = q_mask[b * 512 + q0 + r + 8 * half] ? 1.f : 0.f;
  const unsigned char* kvb = kv_mask + b * 4096;

  float mrow[8], lrow[8];
#pragma unroll
  for (int r = 0; r < 8; r++) { mrow[r] = -INFINITY; lrow[r] = 0.f; }
  v8f oacc[4];
#pragma unroll
  for (int ni = 0; ni < 4; ni++) oacc[ni] = {};

  const unsigned short* Kgb = k_ws + (((size_t)bh * 4096) << 6);
  const unsigned short* Vgb = v_ws + ((size_t)bh * 64) * 4096;
  unsigned short* Pw = Ps + wid * 16 * LDP;

  const int lrk = tid >> 1, lck = (tid & 1) * 32;   // K fill: 128 rows x 64 d
  const int vd  = tid >> 2, vc  = (tid & 3) * 32;   // Vt fill: 64 rows x 128 keys

  // prolog: key block 0 in flight
#pragma unroll
  for (int i = 0; i < 4; i++) {
    async_b128(Kgb, ((unsigned)lrk * 64 + lck) * 2 + i * 16, &Ks[0][lrk * LDT + lck + i * 8]);
    async_b128(Vgb, ((unsigned)vd * 4096 + vc) * 2 + i * 16, &Vts[0][vd * LDP + vc + i * 8]);
  }

  int p = 0;
  for (int kb0 = 0; kb0 < 4096; kb0 += 128, p ^= 1) {
    wait_async0();
    __syncthreads();
    if (kb0 + 128 < 4096) {  // prefetch next key block
#pragma unroll
      for (int i = 0; i < 4; i++) {
        async_b128(Kgb, ((unsigned)(kb0 + 128 + lrk) * 64 + lck) * 2 + i * 16,
                   &Ks[p ^ 1][lrk * LDT + lck + i * 8]);
        async_b128(Vgb, ((unsigned)vd * 4096 + (kb0 + 128) + vc) * 2 + i * 16,
                   &Vts[p ^ 1][vd * LDP + vc + i * 8]);
      }
    }
    // per-lane kv mask for this block's 8 key subtiles
    float kvmf[8];
#pragma unroll
    for (int st = 0; st < 8; st++)
      kvmf[st] = kvb[kb0 + st * 16 + nl] ? 1.f : 0.f;

    // S = Q K^T over 8 subtiles of 16 keys
    v8f s[8];
#pragma unroll
    for (int st = 0; st < 8; st++) {
      v16bf k0f = load_b_frag(Ks[p], st * 16, 0, lane, LDT);
      v16bf k1f = load_b_frag(Ks[p], st * 16, 32, lane, LDT);
      v8f t = {};
      t = __builtin_amdgcn_wmma_f32_16x16x32_bf16(false, qf[0], false, k0f, (short)0, t, false, false);
      t = __builtin_amdgcn_wmma_f32_16x16x32_bf16(false, qf[1], false, k1f, (short)0, t, false, false);
      s[st] = t;
    }
    // mask + per-row running max
    float tmax[8];
#pragma unroll
    for (int r = 0; r < 8; r++) tmax[r] = -INFINITY;
#pragma unroll
    for (int st = 0; st < 8; st++)
#pragma unroll
      for (int r = 0; r < 8; r++) {
        float val = (kvmf[st] * qm[r] != 0.f) ? s[st][r] : -3.402823466e38f;
        s[st][r] = val;
        tmax[r] = fmaxf(tmax[r], val);
      }
#pragma unroll
    for (int r = 0; r < 8; r++)
#pragma unroll
      for (int off = 1; off < 16; off <<= 1)
        tmax[r] = fmaxf(tmax[r], __shfl_xor(tmax[r], off, 32));

    float alpha[8], psum[8];
#pragma unroll
    for (int r = 0; r < 8; r++) {
      float mn = fmaxf(mrow[r], tmax[r]);
      alpha[r] = __expf(mrow[r] - mn);
      mrow[r] = mn;
      psum[r] = 0.f;
    }
#pragma unroll
    for (int st = 0; st < 8; st++)
#pragma unroll
      for (int r = 0; r < 8; r++) {
        float pv = __expf(s[st][r] - mrow[r]);
        s[st][r] = pv;
        psum[r] += pv;
      }
#pragma unroll
    for (int r = 0; r < 8; r++) {
#pragma unroll
      for (int off = 1; off < 16; off <<= 1)
        psum[r] += __shfl_xor(psum[r], off, 32);
      lrow[r] = lrow[r] * alpha[r] + psum[r];
    }
#pragma unroll
    for (int ni = 0; ni < 4; ni++)
#pragma unroll
      for (int r = 0; r < 8; r++) oacc[ni][r] *= alpha[r];

    // C-layout P -> LDS -> A-layout fragments
#pragma unroll
    for (int st = 0; st < 8; st++)
#pragma unroll
      for (int r = 0; r < 8; r++)
        Pw[(r + 8 * half) * LDP + st * 16 + nl] = bf16s(s[st][r]);
    __syncthreads();

    // O += P V  (4 chunks of 32 keys)
#pragma unroll
    for (int c = 0; c < 4; c++) {
      FragBF f;
      const unsigned short* pp = Pw + nl * LDP + c * 32 + (half << 3);
      f.q[0] = *(const uint4*)(pp);
      f.q[1] = *(const uint4*)(pp + 16);
      v16bf pf = f.v;
#pragma unroll
      for (int ni = 0; ni < 4; ni++) {
        v16bf vf = load_b_frag(Vts[p], ni * 16, c * 32, lane, LDP);
        oacc[ni] = __builtin_amdgcn_wmma_f32_16x16x32_bf16(false, pf, false, vf,
                                                           (short)0, oacc[ni], false, false);
      }
    }
  }

  // normalize + store O (bf16) as [b][i][h*64+d]
#pragma unroll
  for (int ni = 0; ni < 4; ni++)
#pragma unroll
    for (int r = 0; r < 8; r++) {
      int i = q0 + r + 8 * half;
      int d = ni * 16 + nl;
      float v = oacc[ni][r] / lrow[r];
      o_ws[((size_t)b * 512 + i) * 1024 + (h << 6) + d] = bf16s(v);
    }
}

// ---------------------------------------------------------------------------
// Host launcher
// ---------------------------------------------------------------------------
extern "C" void kernel_launch(void* const* d_in, const int* in_sizes, int n_in,
                              void* d_out, int out_size, void* d_ws, size_t ws_size,
                              hipStream_t stream) {
  (void)in_sizes; (void)n_in; (void)out_size; (void)ws_size;
  const float* txt   = (const float*)d_in[0];
  const float* image = (const float*)d_in[1];
  const unsigned char* kv_mask = (const unsigned char*)d_in[2];
  const unsigned char* q_mask  = (const unsigned char*)d_in[3];
  const float* Wq   = (const float*)d_in[4];
  const float* Wkv  = (const float*)d_in[5];
  const float* Wout = (const float*)d_in[6];
  const float* bout = (const float*)d_in[7];
  float* out = (float*)d_out;

  char* ws = (char*)d_ws;
  unsigned short* WqT   = (unsigned short*)ws; ws += (size_t)1024 * 1024 * 2;
  unsigned short* WkvT  = (unsigned short*)ws; ws += (size_t)2048 * 1024 * 2;
  unsigned short* WoutT = (unsigned short*)ws; ws += (size_t)1024 * 1024 * 2;
  unsigned short* txt16 = (unsigned short*)ws; ws += (size_t)8 * 512 * 1024 * 2;
  unsigned short* img16 = (unsigned short*)ws; ws += (size_t)8 * 4096 * 1024 * 2;
  unsigned short* q_ws  = (unsigned short*)ws; ws += (size_t)8 * 16 * 512 * 64 * 2;
  unsigned short* k_ws  = (unsigned short*)ws; ws += (size_t)8 * 16 * 4096 * 64 * 2;
  unsigned short* v_ws  = (unsigned short*)ws; ws += (size_t)8 * 16 * 4096 * 64 * 2;
  unsigned short* o_ws  = (unsigned short*)ws; ws += (size_t)8 * 512 * 1024 * 2;

  transpose_cvt_kernel<<<dim3(1024 / 32, 1024 / 32), 256, 0, stream>>>(Wq,   WqT,   1024, 1024);
  transpose_cvt_kernel<<<dim3(2048 / 32, 1024 / 32), 256, 0, stream>>>(Wkv,  WkvT,  1024, 2048);
  transpose_cvt_kernel<<<dim3(1024 / 32, 1024 / 32), 256, 0, stream>>>(Wout, WoutT, 1024, 1024);
  cvt_bf16_kernel<<<dim3((8 * 512 * 1024) / (256 * 8)),  256, 0, stream>>>(txt,   txt16);
  cvt_bf16_kernel<<<dim3((8 * 4096 * 1024) / (256 * 8)), 256, 0, stream>>>(image, img16);

  gemm_q_kernel <<<dim3(8, 16),   256, 0, stream>>>(txt16, WqT,  q_ws);
  gemm_kv_kernel<<<dim3(16, 128), 256, 0, stream>>>(img16, WkvT, k_ws, v_ws);
  attn_kernel   <<<dim3(512),     256, 0, stream>>>(q_ws, k_ws, v_ws, q_mask, kv_mask, o_ws);
  gemm_out_kernel<<<dim3(8, 16),  256, 0, stream>>>(o_ws, WoutT, bout, out);
}
